// LigerChunkedLinearPreferenceBase_44899588112976
// MI455X (gfx1250) — matrix operand order repaired and checked by reading, not compiled, not run on hardware
//
#include <hip/hip_runtime.h>

// ---------------- problem constants ----------------
#define T_TOKENS   4096          // B*S = 8*512
#define SEQ_LEN    512
#define NSEQ       8
#define HDIM       4096
#define VOCAB      32000
#define IGNORE_IDX (-100)
#define BETA       0.1f

// ---------------- tiling ----------------
#define BM   128                 // tokens per WG
#define BN   128                 // vocab rows per WG
#define BK   32                  // K per WMMA step (bf16)
#define LDK  40                  // padded LDS k-stride (bf16): 80B rows (16B aligned)
#define LDF  132                 // padded f32 logits stride
#define NKT  (HDIM / BK)         // 128 (even -> clean 2x unroll)
#define NCHUNK (VOCAB / BN)      // 250
#define NTB    (T_TOKENS / BM)   // 32

#define STAGE_BYTES 20480        // sA(10240) + sB(10240) per stage
#define SB_OFF      10240

typedef __attribute__((ext_vector_type(16))) __bf16 v16bf;
typedef __attribute__((ext_vector_type(8)))  __bf16 v8bf;
typedef __attribute__((ext_vector_type(8)))  float  v8f;

typedef __attribute__((ext_vector_type(4))) unsigned tdm_g0_t;
typedef __attribute__((ext_vector_type(8))) int      tdm_g1_t;
typedef __attribute__((ext_vector_type(4))) int      tdm_g2_t;

union AFrag { v16bf v; v8bf h[2]; };

#if __has_builtin(__builtin_amdgcn_tensor_load_to_lds)
#define USE_TDM 1
#else
#define USE_TDM 0
#endif

// f32 -> bf16 round-to-nearest-even, two packed into a uint (conversion kernel only).
__device__ __forceinline__ unsigned pack_bf16x2(float a, float b) {
    union { float f; unsigned u; } ua, ub;
    ua.f = a; ub.f = b;
    unsigned ra = (ua.u + 0x7FFFu + ((ua.u >> 16) & 1u)) >> 16;
    unsigned rb = (ub.u + 0x7FFFu + ((ub.u >> 16) & 1u)) >> 16;
    return (ra & 0xFFFFu) | (rb << 16);
}

#if USE_TDM
// One TDM tile DMA: 128 rows x 32 bf16 from a row-major (rows x HDIM) bf16
// tensor into LDS at ldsByteOff; pad +16B after every 64B row so the LDS image
// has the 80B (LDK=40) row stride the WMMA fragment loads expect.
__device__ __forceinline__ void tdm_load_tile(unsigned ldsByteOff,
                                              const __bf16* tileStart,
                                              unsigned tensorRows) {
    unsigned long long ga = (unsigned long long)(uintptr_t)tileStart;
    tdm_g0_t g0;
    g0[0] = 1u;                                            // count=1, user mode
    g0[1] = ldsByteOff;                                    // lds_addr (bytes)
    g0[2] = (unsigned)(ga & 0xFFFFFFFFu);                  // global_addr[31:0]
    g0[3] = (unsigned)((ga >> 32) & 0x1FFFFFFu)            // global_addr[56:32]
          | (2u << 30);                                    // type = 2 (image)
    tdm_g1_t g1;
    g1[0] = (int)((1u << 16)                               // data_size = 2B
                | (1u << 20)                               // pad_enable
                | (3u << 22)                               // pad_interval: 16 DW (64B)
                | (3u << 25));                             // pad_amount: 4 DW (16B)
    g1[1] = (int)(((unsigned)HDIM & 0xFFFFu) << 16);       // tensor_dim0[15:0]
    g1[2] = (int)((((unsigned)HDIM >> 16) & 0xFFFFu)       // tensor_dim0[31:16]
                | ((tensorRows & 0xFFFFu) << 16));         // tensor_dim1[15:0]
    g1[3] = (int)(((tensorRows >> 16) & 0xFFFFu)           // tensor_dim1[31:16]
                | ((unsigned)BK << 16));                   // tile_dim0 = 32
    g1[4] = (int)(unsigned)BM;                             // tile_dim1 = 128
    g1[5] = (int)(unsigned)HDIM;                           // tensor_dim0_stride lo
    g1[6] = 0;
    g1[7] = 0;
    tdm_g2_t z4 = {0, 0, 0, 0};
#if __clang_major__ >= 23
    tdm_g1_t z8 = {0, 0, 0, 0, 0, 0, 0, 0};
    __builtin_amdgcn_tensor_load_to_lds(g0, g1, z4, z4, z8, 0);
#else
    __builtin_amdgcn_tensor_load_to_lds(g0, g1, z4, z4, 0);
#endif
}
#endif // USE_TDM

// ---------------------------------------------------------------------------
// Kernel 0: one-time f32 -> bf16 conversion (8 elements / thread).
// ---------------------------------------------------------------------------
__global__ __launch_bounds__(256)
void convert_f32_to_bf16(const float* __restrict__ in,
                         uint4* __restrict__ out, long long n8) {
    long long i = (long long)blockIdx.x * blockDim.x + threadIdx.x;
    if (i >= n8) return;
    const float4* p = (const float4*)in + i * 2;
    float4 a = p[0], b = p[1];
    uint4 o;
    o.x = pack_bf16x2(a.x, a.y);
    o.y = pack_bf16x2(a.z, a.w);
    o.z = pack_bf16x2(b.x, b.y);
    o.w = pack_bf16x2(b.z, b.w);
    out[i] = o;
}

// ---------------------------------------------------------------------------
// Kernel 1: fused bf16-WMMA GEMM tile + online softmax partials.
// grid.x = token block (fast-varying -> consecutive WGs share the W tile in L2)
// grid.y = vocab chunk
// ---------------------------------------------------------------------------
__global__ __launch_bounds__(256, 2)   // 8 waves / 4 SIMDs -> only 2 waves/SIMD needed
void dpo_fused_gemm_stats(const __bf16* __restrict__ xbf,
                          const __bf16* __restrict__ wbf,
                          const int*    __restrict__ target,
                          float4*       __restrict__ stats) {
    // LDS: 2 stages x (sA 10240B + sB 10240B) = 40960B, union'd with the
    // 128x132 f32 logits tile (67584B) used only after the K loop.
    __shared__ __attribute__((aligned(16))) unsigned char smem[BM * LDF * 4];

    const int tokenBase = blockIdx.x * BM;
    const int chunkBase = blockIdx.y * BN;

    const int tid   = threadIdx.x;
    const int lane  = tid & 31;
    const int wave  = tid >> 5;          // 8 waves
    const int waveM = wave & 1;          // 2 waves over M (64 rows each)
    const int waveN = wave >> 1;         // 4 waves over N (32 cols each)
    const int l15   = lane & 15;
    const bool hi   = lane >= 16;
    const bool isWave0 = (wave == 0);

    const __bf16* aTile0 = xbf + (size_t)tokenBase * HDIM;
    const __bf16* bTile0 = wbf + (size_t)chunkBase * HDIM;

    const v8f vzero = {0.f, 0.f, 0.f, 0.f, 0.f, 0.f, 0.f, 0.f};
    v8f acc[4][2];
    #pragma unroll
    for (int mt = 0; mt < 4; ++mt)
        #pragma unroll
        for (int nt = 0; nt < 2; ++nt) acc[mt][nt] = vzero;

#if USE_TDM
    auto issueStage = [&](int kt, int buf) {
        tdm_load_tile((unsigned)(buf * STAGE_BYTES),          aTile0 + kt * BK, T_TOKENS);
        tdm_load_tile((unsigned)(buf * STAGE_BYTES + SB_OFF), bTile0 + kt * BK, VOCAB);
    };
    if (isWave0) {
        issueStage(0, 0);
        __builtin_amdgcn_s_wait_tensorcnt((short)0);
    }
    __syncthreads();
#else
    auto copyStage = [&](int kt, int buf) {
        __bf16* sA = (__bf16*)(smem + buf * STAGE_BYTES);
        __bf16* sB = (__bf16*)(smem + buf * STAGE_BYTES + SB_OFF);
        #pragma unroll
        for (int j = 0; j < 2; ++j) {
            int i = tid + 256 * j;       // uint4 index 0..511, 4 per row
            int r = i >> 2, c = i & 3;
            *(uint4*)&sA[r * LDK + c * 8] =
                *(const uint4*)(aTile0 + (size_t)r * HDIM + kt * BK + c * 8);
            *(uint4*)&sB[r * LDK + c * 8] =
                *(const uint4*)(bTile0 + (size_t)r * HDIM + kt * BK + c * 8);
        }
    };
    copyStage(0, 0);
    __syncthreads();
#endif

    // One MMA step from a compile-time stage; constant DS offsets.
    auto computeStage = [&](int buf) {
        const __bf16* sA = (const __bf16*)(smem + buf * STAGE_BYTES);
        const __bf16* sB = (const __bf16*)(smem + buf * STAGE_BYTES + SB_OFF);

        // B fragments (32x16): lane = column, K = 8h+s / 16+8h+(s-8) per lane; reused 4x.
        AFrag b0, b1;
        {
            int n0 = waveN * 32 + l15;
            const __bf16* p0 = sB + n0 * LDK + (hi ? 8 : 0);
            b0.h[0] = *(const v8bf*)p0;
            b0.h[1] = *(const v8bf*)(p0 + 16);
            const __bf16* p1 = p0 + 16 * LDK;
            b1.h[0] = *(const v8bf*)p1;
            b1.h[1] = *(const v8bf*)(p1 + 16);
        }
        // Per M-tile: one A fragment (8 VGPRs) feeds its two WMMAs immediately.
        #pragma unroll
        for (int mt = 0; mt < 4; ++mt) {
            AFrag a;
            int m = waveM * 64 + mt * 16 + l15;
            const __bf16* p = sA + m * LDK + (hi ? 8 : 0);
            a.h[0] = *(const v8bf*)p;
            a.h[1] = *(const v8bf*)(p + 16);
            acc[mt][0] = __builtin_amdgcn_wmma_f32_16x16x32_bf16(
                false, a.v, false, b0.v, (short)0, acc[mt][0], false, false);
            acc[mt][1] = __builtin_amdgcn_wmma_f32_16x16x32_bf16(
                false, a.v, false, b1.v, (short)0, acc[mt][1], false, false);
        }
    };

    // Manually 2x-unrolled ping-pong so both stage addresses are immediates.
    for (int kt2 = 0; kt2 < NKT; kt2 += 2) {
        // ---- even step: compute stage 0, DMA (kt2+1) into stage 1 ----
#if USE_TDM
        if (isWave0) issueStage(kt2 + 1, 1);          // kt2+1 < NKT (NKT even)
#else
        copyStage(kt2 + 1, 1);
#endif
        computeStage(0);
#if USE_TDM
        if (isWave0) __builtin_amdgcn_s_wait_tensorcnt((short)0);
#endif
        __syncthreads();

        // ---- odd step: compute stage 1, DMA (kt2+2) into stage 0 ----
#if USE_TDM
        if (isWave0 && kt2 + 2 < NKT) issueStage(kt2 + 2, 0);
#else
        if (kt2 + 2 < NKT) copyStage(kt2 + 2, 0);
#endif
        computeStage(1);
#if USE_TDM
        if (isWave0 && kt2 + 2 < NKT)
            __builtin_amdgcn_s_wait_tensorcnt((short)0);
#endif
        __syncthreads();
    }

    // ----- epilogue: spill 128x128 f32 logits into LDS (reusing staging space) -----
    float* Lg = (float*)smem;
    #pragma unroll
    for (int mt = 0; mt < 4; ++mt) {
        #pragma unroll
        for (int nt = 0; nt < 2; ++nt) {
            int mBase = waveM * 64 + mt * 16 + (hi ? 8 : 0);
            int col   = waveN * 32 + nt * 16 + l15;
            #pragma unroll
            for (int r = 0; r < 8; ++r)
                Lg[(mBase + r) * LDF + col] = acc[mt][nt][r];
        }
    }
    __syncthreads();

    // ----- per-row online-softmax partials over this 128-wide vocab chunk -----
    if (tid < BM) {
        const float* Lr = Lg + tid * LDF;
        float m = -3.0e38f;
        for (int j = 0; j < BN; ++j) m = fmaxf(m, Lr[j]);
        float sexp = 0.f, ssum = 0.f;
        for (int j = 0; j < BN; ++j) {
            float v = Lr[j];
            sexp += __expf(v - m);
            ssum += v;
        }
        int tok = tokenBase + tid;
        int tgt = target[tok];
        int local = tgt - chunkBase;
        float tl = (local >= 0 && local < BN) ? Lr[local] : 0.f;
        float4 st; st.x = m; st.y = sexp; st.z = ssum; st.w = tl;
        stats[(size_t)blockIdx.y * T_TOKENS + tok] = st;
    }
}

// ---------------------------------------------------------------------------
// Kernel 2: merge online-softmax partials across the 250 vocab chunks.
// ---------------------------------------------------------------------------
__global__ __launch_bounds__(256)
void dpo_merge_tokens(const float4* __restrict__ stats,
                      const int*    __restrict__ target,
                      float* __restrict__ tokLogp,
                      float* __restrict__ tokMask,
                      float* __restrict__ tokSum) {
    int t = blockIdx.x * blockDim.x + threadIdx.x;
    if (t >= T_TOKENS) return;
    float m = -3.0e38f, s = 0.f, sum = 0.f, tl = 0.f;
    for (int c = 0; c < NCHUNK; ++c) {
        float4 st = stats[(size_t)c * T_TOKENS + t];
        float mo = m;
        m = fmaxf(m, st.x);
        s = s * __expf(mo - m) + st.y * __expf(st.x - m);
        sum += st.z;
        tl  += st.w;
    }
    float logp = tl - (m + __logf(s));
    bool mask  = (target[t] != IGNORE_IDX);
    tokLogp[t] = mask ? logp : 0.f;
    tokMask[t] = mask ? 1.f : 0.f;
    tokSum[t]  = sum;
}

// ---------------------------------------------------------------------------
// Kernel 3: per-sequence reductions + DPO loss -> 12 output floats.
// ---------------------------------------------------------------------------
__global__ void dpo_finalize(const float* __restrict__ tokLogp,
                             const float* __restrict__ tokMask,
                             const float* __restrict__ tokSum,
                             float* __restrict__ out) {
    __shared__ float seqLogp[NSEQ];
    __shared__ float seqSum[NSEQ];
    int b = threadIdx.x;
    if (b < NSEQ) {
        float sl = 0.f, sm = 0.f, ss = 0.f;
        for (int s = 0; s < SEQ_LEN; ++s) {
            int t = b * SEQ_LEN + s;
            sl += tokLogp[t];
            sm += tokMask[t];
            ss += tokSum[t];
        }
        seqLogp[b] = sl / fmaxf(sm, 1.f);
        seqSum[b]  = ss;
    }
    __syncthreads();
    if (threadIdx.x == 0) {
        float loss = 0.f, csum = 0.f, rsum = 0.f;
        for (int i = 0; i < NSEQ / 2; ++i) {
            float d = BETA * (seqLogp[i] - seqLogp[i + NSEQ / 2]);
            loss += __logf(1.f + __expf(-d));   // -log_sigmoid(d)
            csum += seqSum[i];
            rsum += seqSum[i + NSEQ / 2];
            out[1 + i] = seqLogp[i];
            out[5 + i] = seqLogp[i + NSEQ / 2];
        }
        const float denom = (float)(NSEQ / 2) * (float)SEQ_LEN * (float)VOCAB;
        out[0]  = loss / (float)(NSEQ / 2);
        out[9]  = csum / denom;
        out[10] = rsum / denom;
        out[11] = 0.f;                          // nll_loss
    }
}

// ---------------------------------------------------------------------------
extern "C" void kernel_launch(void* const* d_in, const int* in_sizes, int n_in,
                              void* d_out, int out_size, void* d_ws, size_t ws_size,
                              hipStream_t stream) {
    (void)in_sizes; (void)n_in; (void)out_size; (void)ws_size;
    const float* x      = (const float*)d_in[0];   // (8,512,4096) f32
    const float* w      = (const float*)d_in[1];   // (32000,4096) f32
    const int*   target = (const int*)  d_in[2];   // (8,512) i32
    float* out = (float*)d_out;                    // 12 floats

    // workspace layout (bytes)
    const long long nW = (long long)VOCAB * HDIM;       // 131,072,000
    const long long nX = (long long)T_TOKENS * HDIM;    //  16,777,216
    unsigned char* wsb = (unsigned char*)d_ws;
    __bf16* wbf   = (__bf16*)wsb;                                   // 262,144,000B
    __bf16* xbf   = (__bf16*)(wsb + nW * 2);                        //  33,554,432B
    float4* stats = (float4*)(wsb + nW * 2 + nX * 2);               //  16,384,000B
    float*  tokLogp = (float*)(stats + (size_t)NCHUNK * T_TOKENS);
    float*  tokMask = tokLogp + T_TOKENS;
    float*  tokSum  = tokMask + T_TOKENS;

    // one-time precision conversion (removes all cvt work + halves W traffic)
    const long long w8 = nW / 8, x8 = nX / 8;
    convert_f32_to_bf16<<<(unsigned)((w8 + 255) / 256), 256, 0, stream>>>(
        w, (uint4*)wbf, w8);
    convert_f32_to_bf16<<<(unsigned)((x8 + 255) / 256), 256, 0, stream>>>(
        x, (uint4*)xbf, x8);

    dim3 grid1(NTB, NCHUNK);      // token block fast-varying -> W-tile L2 reuse
    dpo_fused_gemm_stats<<<grid1, 256, 0, stream>>>(xbf, wbf, target, stats);

    dpo_merge_tokens<<<(T_TOKENS + 255) / 256, 256, 0, stream>>>(
        stats, target, tokLogp, tokMask, tokSum);

    dpo_finalize<<<1, 64, 0, stream>>>(tokLogp, tokMask, tokSum, out);
}
